// TreeLSTM_29128468201683
// MI455X (gfx1250) — matrix operand order repaired and checked by reading, not compiled, not run on hardware
//
#include <hip/hip_runtime.h>

#define XH   512
#define NPER 5461
#define NTOT 87376
#define BPAD 520                    // LDS row stride (elements): 1040 B -> conflict-free frags
#define LDSB (64 * BPAD * 2)        // 66560 B dynamic LDS per block

typedef __attribute__((ext_vector_type(16))) __bf16 v16bf;
typedef __attribute__((ext_vector_type(8)))  __bf16 v8bf;
typedef __attribute__((ext_vector_type(8)))  float  v8f;

__device__ __forceinline__ float sigm(float v) { return 1.0f / (1.0f + __expf(-v)); }

// concatenated-level row r -> global node id: (r / 4^d)*NPER + lo_d + (r % 4^d)
__device__ __forceinline__ size_t rowmap(int r, int lo, int shift, int mask) {
  return (size_t)((unsigned)r >> shift) * (size_t)NPER + (size_t)lo + (size_t)(r & mask);
}

__global__ __launch_bounds__(256) void cvt_bf16_kernel(const float* __restrict__ s,
                                                       __bf16* __restrict__ d, int n) {
  int i = blockIdx.x * 256 + threadIdx.x;
  if (i < n) d[i] = (__bf16)s[i];
}

// C[r, c] = act( sum_k A[rowmap(r0+r)][k] * W[c][k] + bias[c] ),  K = 512.
// A is bf16 row-major [*,512]; W is bf16 row-major [cols,512].
// Block: 8 waves -> 128 rows x 64 cols; B panel staged in LDS once (contiguous slab of W).
__global__ __launch_bounds__(256) void gemm_bf16_kernel(
    const __bf16* __restrict__ A, int lo, int shift, int mask, int r0,
    const __bf16* __restrict__ W, const float* __restrict__ bias,
    float* __restrict__ C, int ldC, int rows, int cols, int act)
{
  extern __shared__ __bf16 Bs[];
  const int tid  = threadIdx.x;
  const int col0 = blockIdx.x * 64;

  // Stage 64x512 bf16 weight panel -> LDS (padded rows). 4096 chunks of 16 B.
  {
    const __bf16* src = W + (size_t)col0 * XH;
    #pragma unroll
    for (int j = 0; j < 16; ++j) {
      const int cid = tid + j * 256;
      const int row = cid >> 6;            // 64 chunks per 512-elem row
      const int off = (cid & 63) * 8;
      *(v8bf*)(Bs + row * BPAD + off) = *(const v8bf*)(src + (size_t)row * XH + off);
    }
  }
  __syncthreads();

  const int wave = tid >> 5;
  const int lane = tid & 31;
  const int hf   = lane >> 4;              // lane half = K-phase (ISA 16-bit A/B layouts)
  const int l16  = lane & 15;
  const int row_t = blockIdx.y * 128 + wave * 16;
  if (row_t >= rows) return;               // wave-uniform: EXEC stays all-ones

  const int rr = r0 + row_t + l16;
  const __bf16* __restrict__ arow = A + rowmap(rr, lo, shift, mask) * XH;

  v8f acc[4] = {};

  #pragma unroll 4
  for (int k0 = 0; k0 < XH; k0 += 32) {
    // A fragment: row l16, K {k0+8hf..+8, k0+16+8hf..+8} -> two 16 B global loads
    const v8bf a0 = *(const v8bf*)(arow + k0 + 8 * hf);
    const v8bf a1 = *(const v8bf*)(arow + k0 + 16 + 8 * hf);
    v16bf a;
    #pragma unroll
    for (int j = 0; j < 8; ++j) { a[j] = a0[j]; a[8 + j] = a1[j]; }

    // Preload all four B fragments of this k-step so the WMMA quartet issues
    // back-to-back behind a single dscnt wait.
    v16bf bfr[4];
    #pragma unroll
    for (int nt = 0; nt < 4; ++nt) {
      const __bf16* bp = Bs + (16 * nt + l16) * BPAD + k0 + 8 * hf;
      const v8bf b0 = *(const v8bf*)bp;
      const v8bf b1 = *(const v8bf*)(bp + 16);
      #pragma unroll
      for (int j = 0; j < 8; ++j) { bfr[nt][j] = b0[j]; bfr[nt][8 + j] = b1[j]; }
    }
    #pragma unroll
    for (int nt = 0; nt < 4; ++nt) {
      acc[nt] = __builtin_amdgcn_wmma_f32_16x16x32_bf16(
          false, a, false, bfr[nt], (short)0, acc[nt], false, false);
    }
  }

  // C/D layout: VGPR r -> M = r + 8*hf, N = l16
  #pragma unroll
  for (int nt = 0; nt < 4; ++nt) {
    #pragma unroll
    for (int r = 0; r < 8; ++r) {
      const int orow = row_t + r + 8 * hf;
      const int ocol = col0 + 16 * nt + l16;
      float v = acc[nt][r] + bias[ocol];
      if (act) v = sigm(v);
      C[(size_t)orow * ldC + ocol] = v;
    }
  }
}

// Node chunk [n0,n0+cnt) at level d: htild_bf[q] = sum_k h[child], c[node] = sum_k f*c[child]
__global__ __launch_bounds__(256) void reduce_kernel(
    const float* __restrict__ G, const __bf16* __restrict__ hbf, float* c,
    __bf16* __restrict__ htild_bf,
    int loD, int shD, int mkD, int loC, int shC, int mkC, int n0, int cnt)
{
  int idx = blockIdx.x * 256 + threadIdx.x;
  if (idx >= cnt * 512) return;
  int q = idx >> 9, col = idx & 511;
  int rn = n0 + q;
  size_t gn = rowmap(rn, loD, shD, mkD);
  float hs = 0.f, cs = 0.f;
  #pragma unroll
  for (int k = 0; k < 4; ++k) {            // children of concat-row rn are concat-rows 4rn+k
    size_t gc = rowmap(4 * rn + k, loC, shC, mkC);
    hs += (float)hbf[gc * XH + col];
    cs += G[(size_t)(4 * q + k) * XH + col] * c[gc * XH + col];
  }
  htild_bf[(size_t)q * XH + col] = (__bf16)hs;
  c[gn * XH + col] = cs;                   // c_in; completed by elemwise_kernel
}

__global__ __launch_bounds__(256) void elemwise_kernel(
    const float* __restrict__ iou, int lo, int shift, int mask, int r0,
    __bf16* __restrict__ hbf, float* __restrict__ c, int rows, int has_cin)
{
  int idx = blockIdx.x * 256 + threadIdx.x;
  if (idx >= rows * 512) return;
  int r = idx >> 9, col = idx & 511;
  size_t g = rowmap(r0 + r, lo, shift, mask);
  float iv = iou[(size_t)r * 1536 + col];
  float ov = iou[(size_t)r * 1536 + 512 + col];
  float uv = iou[(size_t)r * 1536 + 1024 + col];
  float cin = has_cin ? c[g * XH + col] : 0.f;
  float cn = sigm(iv) * tanhf(uv) + cin;
  c[g * XH + col] = cn;
  hbf[g * XH + col] = (__bf16)(sigm(ov) * tanhf(cn));
}

extern "C" void kernel_launch(void* const* d_in, const int* in_sizes, int n_in,
                              void* d_out, int out_size, void* d_ws, size_t ws_size,
                              hipStream_t stream)
{
  const float* x     = (const float*)d_in[0];
  const float* W_iou = (const float*)d_in[1];
  const float* U_iou = (const float*)d_in[2];
  const float* b_iou = (const float*)d_in[3];
  const float* U_f_w = (const float*)d_in[4];
  const float* U_f_b = (const float*)d_in[5];
  const float* lin_w = (const float*)d_in[6];
  const float* lin_b = (const float*)d_in[7];

  char* w = (char*)d_ws;
  float*  c        = (float*)w;   w += (size_t)NTOT * 512 * 4;
  __bf16* hbf      = (__bf16*)w;  w += (size_t)NTOT * 512 * 2;
  __bf16* xbf      = (__bf16*)w;  w += (size_t)NTOT * 512 * 2;
  __bf16* Wiou_bf  = (__bf16*)w;  w += (size_t)1536 * 512 * 2;
  __bf16* Uiou_bf  = (__bf16*)w;  w += (size_t)1536 * 512 * 2;
  __bf16* Ufw_bf   = (__bf16*)w;  w += (size_t)512 * 512 * 2;
  __bf16* Lin_bf   = (__bf16*)w;  w += (size_t)512 * 512 * 2;
  float*  G        = (float*)w;   w += (size_t)16384 * 512 * 4;    // forget gates (child rows)
  __bf16* htild_bf = (__bf16*)w;  w += (size_t)4096 * 512 * 2;     // child-sum h
  float*  iou      = (float*)w;   w += (size_t)16384 * 1536 * 4;   // iou scratch

  cvt_bf16_kernel<<<(1536 * 512 + 255) / 256, 256, 0, stream>>>(W_iou, Wiou_bf, 1536 * 512);
  cvt_bf16_kernel<<<(1536 * 512 + 255) / 256, 256, 0, stream>>>(U_iou, Uiou_bf, 1536 * 512);
  cvt_bf16_kernel<<<(512 * 512 + 255) / 256, 256, 0, stream>>>(U_f_w, Ufw_bf, 512 * 512);
  cvt_bf16_kernel<<<(512 * 512 + 255) / 256, 256, 0, stream>>>(lin_w, Lin_bf, 512 * 512);
  cvt_bf16_kernel<<<(NTOT * 512 + 255) / 256, 256, 0, stream>>>(x, xbf, NTOT * 512);

  static const int LO[7] = {0, 1, 5, 21, 85, 341, 1365};  // (4^d - 1)/3

  // Leaves (depth 6): iou = x @ W_iou.T + b_iou ; gates with c_in = 0. 65536 rows in 4 chunks.
  for (int ch = 0; ch < 4; ++ch) {
    int r0 = ch * 16384, rows = 16384;
    gemm_bf16_kernel<<<dim3(1536 / 64, rows / 128), 256, LDSB, stream>>>(
        xbf, LO[6], 12, 4095, r0, Wiou_bf, b_iou, iou, 1536, rows, 1536, 0);
    elemwise_kernel<<<rows * 512 / 256, 256, 0, stream>>>(
        iou, LO[6], 12, 4095, r0, hbf, c, rows, 0);
  }

  // Internal levels, bottom-up.
  for (int d = 5; d >= 0; --d) {
    int ntot = 16 << (2 * d);
    int CN   = ntot > 4096 ? 4096 : ntot;
    int loD = LO[d],     shD = 2 * d,       mkD = (1 << (2 * d)) - 1;
    int loC = LO[d + 1], shC = 2 * (d + 1), mkC = (1 << (2 * (d + 1))) - 1;
    for (int n0 = 0; n0 < ntot; n0 += CN) {
      int cnt = CN, crows = 4 * cnt;
      // f = sigmoid(h_child @ U_f_w.T + U_f_b) over the contiguous child block
      gemm_bf16_kernel<<<dim3(512 / 64, (crows + 127) / 128), 256, LDSB, stream>>>(
          hbf, loC, shC, mkC, 4 * n0, Ufw_bf, U_f_b, G, 512, crows, 512, 1);
      reduce_kernel<<<(cnt * 512 + 255) / 256, 256, 0, stream>>>(
          G, hbf, c, htild_bf, loD, shD, mkD, loC, shC, mkC, n0, cnt);
      // iou = h_tild @ U_iou.T + b_iou  (overwrite semantics of the reference)
      gemm_bf16_kernel<<<dim3(1536 / 64, (cnt + 127) / 128), 256, LDSB, stream>>>(
          htild_bf, 0, 30, 0x3fffffff, 0, Uiou_bf, b_iou, iou, 1536, cnt, 1536, 0);
      elemwise_kernel<<<(cnt * 512 + 255) / 256, 256, 0, stream>>>(
          iou, loD, shD, mkD, n0, hbf, c, cnt, 1);
    }
  }

  // out = h @ lin_w.T + lin_b over all N nodes
  gemm_bf16_kernel<<<dim3(512 / 64, (NTOT + 127) / 128), 256, LDSB, stream>>>(
      hbf, 0, 30, 0x3fffffff, 0, Lin_bf, lin_b, (float*)d_out, 512, NTOT, 512, 0);
}
